// CINppConv_53085795779158
// MI455X (gfx1250) — compile-verified, matrix operand order checked by reading, not run on hardware
//
#include <hip/hip_runtime.h>

typedef __bf16 bf16;
typedef __attribute__((ext_vector_type(16))) __bf16 v16bf;
typedef __attribute__((ext_vector_type(8)))  __bf16 v8bf;
typedef __attribute__((ext_vector_type(8)))  unsigned v8u;
typedef __attribute__((ext_vector_type(8)))  float  v8f;

constexpr int  N_CELLS = 32768;
constexpr int  D       = 128;
constexpr int  E_UP    = 524288;
constexpr int  E_DN    = 524288;
constexpr int  E_B     = 131072;
constexpr float BN_EPS = 1e-5f;
constexpr float INV_N  = 1.0f / (float)N_CELLS;

// float -> bf16 (round to nearest even), scalar path for cheap kernels
__device__ __forceinline__ bf16 f2bf(float f) {
  unsigned u = __builtin_bit_cast(unsigned, f);
  u += 0x7fffu + ((u >> 16) & 1u);
  unsigned short h = (unsigned short)(u >> 16);
  return __builtin_bit_cast(bf16, h);
}

// Hardware packed conversion: one VALU op for two floats -> packed bf16 pair.
// (CDNA4/5 V_CVT_PK_BF16_F32; no clang builtin declared for gfx1250, so asm.)
__device__ __forceinline__ unsigned cvt_pk_bf16(float a, float b) {
  unsigned r;
  asm("v_cvt_pk_bf16_f32 %0, %1, %2" : "=v"(r) : "v"(a), "v"(b));
  return r;
}

// hardware f32 atomic add (global_atomic_add_f32)
__device__ __forceinline__ void atomAddF(float* p, float v) {
  unsafeAtomicAdd(p, v);
}

// ---------------------------------------------------------------------------
// Weight packing: row-major [K x 128] f32 -> bf16 WMMA B-fragments.
// B-fragment layout (mirror of the ISA-documented 16-bit A layout, lane ->
// column): for a 32x16 (KxN) block, element (kk, nn):
//   grp = kk>>3; lane = nn + 16*(grp&1); j = (kk&7) + 8*(grp>>1)
// Fragment (kt,nt) stored at ((kt*8 + nt)*512 + lane*16 + j).
// blockIdx.y batches independent matrices.
// ---------------------------------------------------------------------------
__global__ void pack_w(const float* __restrict__ src, bf16* __restrict__ dst, int K) {
  int b = blockIdx.y;
  src += (long)b * K * 128;
  dst += (long)b * (K / 32) * 8 * 512;
  int e = blockIdx.x * 256 + threadIdx.x;
  if (e >= K * 128) return;
  int k = e >> 7, n = e & 127;
  int kt = k >> 5, kk = k & 31, nt = n >> 4, nn = n & 15;
  int grp = kk >> 3;
  int lane = nn + ((grp & 1) << 4);
  int j = (kk & 7) + ((grp >> 1) << 3);
  dst[((kt * 8 + nt) << 9) + (lane << 4) + j] = f2bf(src[e]);
}

// acc[k][n][d] = x[n][d] for k=0..2 (folds the "+ x" GIN self term; also the
// zero-init target for the scatter-adds). Also zeroes the BN stats region.
__global__ void init_acc(const float* __restrict__ x, float* __restrict__ acc,
                         float* __restrict__ stats) {
  long g = (long)blockIdx.x * 256 + threadIdx.x;  // over N*D
  float v = x[g];
  acc[g] = v;
  acc[g + (long)N_CELLS * D] = v;
  acc[g + 2L * N_CELLS * D] = v;
  if (g < 1792) stats[g] = 0.f;
}

// Build a 16x32 bf16 A-fragment for this lane from a contiguous f32 row.
// Lane (m, hi): holds row m; K set = {d..d+7} U {d+16..d+23}, d = 8*hi.
__device__ __forceinline__ v16bf load_a_f32(const float* __restrict__ rp) {
  float4 l0 = *(const float4*)(rp);
  float4 l1 = *(const float4*)(rp + 4);
  float4 h0 = *(const float4*)(rp + 16);
  float4 h1 = *(const float4*)(rp + 20);
  v8u u;
  u[0] = cvt_pk_bf16(l0.x, l0.y);
  u[1] = cvt_pk_bf16(l0.z, l0.w);
  u[2] = cvt_pk_bf16(l1.x, l1.y);
  u[3] = cvt_pk_bf16(l1.z, l1.w);
  u[4] = cvt_pk_bf16(h0.x, h0.y);
  u[5] = cvt_pk_bf16(h0.z, h0.w);
  u[6] = cvt_pk_bf16(h1.x, h1.y);
  u[7] = cvt_pk_bf16(h1.z, h1.w);
  return __builtin_bit_cast(v16bf, u);
}

__device__ __forceinline__ v16bf load_a_bf16(const bf16* __restrict__ rp) {
  v8bf lo = *(const v8bf*)(rp);
  v8bf hi = *(const v8bf*)(rp + 16);
  return __builtin_shufflevector(lo, hi, 0,1,2,3,4,5,6,7,8,9,10,11,12,13,14,15);
}

// ---------------------------------------------------------------------------
// Edge-message kernel:  m = ReLU([x[src] ; attr] @ W + b); atomic scatter-add
// to out[dst]. 8 waves/block; wave = 16 edges x 128 cols; K = 256 (8 steps).
// ---------------------------------------------------------------------------
__global__ void __launch_bounds__(256)
msg_kernel(const float* __restrict__ x, const int* __restrict__ idx,
           const float* __restrict__ attr, const bf16* __restrict__ wpack,
           const float* __restrict__ bias, float* __restrict__ out, int E) {
  const int lane = threadIdx.x & 31;
  const int wv   = threadIdx.x >> 5;
  const int e0   = blockIdx.x * 128 + wv * 16;
  const int m    = lane & 15;
  const int hi   = lane >> 4;
  const int dofs = hi << 3;
  const int e    = e0 + m;
  const int src  = idx[E + e];                       // up/down_index[1]
  const float* __restrict__ xrow = x    + (long)src * D;
  const float* __restrict__ arow = attr + (long)e   * D;

  v8f acc[8];
#pragma unroll
  for (int t = 0; t < 8; ++t) acc[t] = v8f{};

#pragma unroll
  for (int ks = 0; ks < 8; ++ks) {
    const int kb = ks * 32;
    const float* rp = (kb < 128) ? (xrow + kb + dofs) : (arow + (kb - 128) + dofs);
    v16bf a = load_a_f32(rp);
#pragma unroll
    for (int nt = 0; nt < 8; ++nt) {
      v16bf b = *(const v16bf*)(wpack + (((ks << 3) + nt) << 9) + (lane << 4));
      acc[nt] = __builtin_amdgcn_wmma_f32_16x16x32_bf16(
          false, a, false, b, (short)0, acc[nt], false, false);
    }
  }

  int drow[8];
#pragma unroll
  for (int r = 0; r < 8; ++r) drow[r] = idx[e0 + hi * 8 + r];  // up/down_index[0]

  const int col0 = lane & 15;
#pragma unroll
  for (int nt = 0; nt < 8; ++nt) {
    const int col = col0 + (nt << 4);
    const float bv = bias[col];
#pragma unroll
    for (int r = 0; r < 8; ++r) {
      float v = acc[nt][r] + bv;
      v = v > 0.f ? v : 0.f;
      atomAddF(out + (long)drow[r] * D + col, v);
    }
  }
}

// boundary: out[dst] += boundary_attr[src]  (pure gather + atomic scatter)
__global__ void boundary_kernel(const int* __restrict__ idx,
                                const float* __restrict__ attr,
                                float* __restrict__ out) {
  long g = (long)blockIdx.x * 256 + threadIdx.x;   // E_B * 32 threads
  int e  = (int)(g >> 5);
  int cg = ((int)g & 31) << 2;
  int srow = idx[e];
  int drow = idx[E_B + e];
  float4 v = *(const float4*)(attr + (long)srow * D + cg);
  float* o = out + (long)drow * D + cg;
  atomAddF(o + 0, v.x); atomAddF(o + 1, v.y);
  atomAddF(o + 2, v.z); atomAddF(o + 3, v.w);
}

// ---------------------------------------------------------------------------
// Dense GEMMs: Z[br] = A[br] @ W[br] + b[br], A is [N x 128], K = 128.
// blockIdx.y = branch. Wave = 16 rows x 128 cols.
// ---------------------------------------------------------------------------
__global__ void __launch_bounds__(256)
gemm_f32A(const float* __restrict__ A, const bf16* __restrict__ wpack,
          const float* __restrict__ bias, float* __restrict__ Z) {
  const int br = blockIdx.y;
  A     += (long)br * N_CELLS * D;
  Z     += (long)br * N_CELLS * D;
  wpack += (long)br * 4 * 8 * 512;
  bias  += br * D;
  const int lane = threadIdx.x & 31;
  const int wv   = threadIdx.x >> 5;
  const int r0   = blockIdx.x * 128 + wv * 16;
  const int m    = lane & 15;
  const int hi   = lane >> 4;
  const int dofs = hi << 3;
  const float* __restrict__ arow = A + (long)(r0 + m) * D;

  v8f acc[8];
#pragma unroll
  for (int t = 0; t < 8; ++t) acc[t] = v8f{};
#pragma unroll
  for (int ks = 0; ks < 4; ++ks) {
    v16bf a = load_a_f32(arow + ks * 32 + dofs);
#pragma unroll
    for (int nt = 0; nt < 8; ++nt) {
      v16bf b = *(const v16bf*)(wpack + (((ks << 3) + nt) << 9) + (lane << 4));
      acc[nt] = __builtin_amdgcn_wmma_f32_16x16x32_bf16(
          false, a, false, b, (short)0, acc[nt], false, false);
    }
  }
  const int col0 = lane & 15;
#pragma unroll
  for (int nt = 0; nt < 8; ++nt) {
    int col = col0 + (nt << 4);
    float bv = bias[col];
#pragma unroll
    for (int r = 0; r < 8; ++r)
      Z[(long)(r0 + hi * 8 + r) * D + col] = acc[nt][r] + bv;
  }
}

__global__ void __launch_bounds__(256)
gemm_bf16A(const bf16* __restrict__ A, const bf16* __restrict__ wpack,
           const float* __restrict__ bias, float* __restrict__ Z) {
  const int br = blockIdx.y;
  A     += (long)br * N_CELLS * D;
  Z     += (long)br * N_CELLS * D;
  wpack += (long)br * 4 * 8 * 512;
  bias  += br * D;
  const int lane = threadIdx.x & 31;
  const int wv   = threadIdx.x >> 5;
  const int r0   = blockIdx.x * 128 + wv * 16;
  const int m    = lane & 15;
  const int hi   = lane >> 4;
  const int dofs = hi << 3;
  const bf16* __restrict__ arow = A + (long)(r0 + m) * D;

  v8f acc[8];
#pragma unroll
  for (int t = 0; t < 8; ++t) acc[t] = v8f{};
#pragma unroll
  for (int ks = 0; ks < 4; ++ks) {
    v16bf a = load_a_bf16(arow + ks * 32 + dofs);
#pragma unroll
    for (int nt = 0; nt < 8; ++nt) {
      v16bf b = *(const v16bf*)(wpack + (((ks << 3) + nt) << 9) + (lane << 4));
      acc[nt] = __builtin_amdgcn_wmma_f32_16x16x32_bf16(
          false, a, false, b, (short)0, acc[nt], false, false);
    }
  }
  const int col0 = lane & 15;
#pragma unroll
  for (int nt = 0; nt < 8; ++nt) {
    int col = col0 + (nt << 4);
    float bv = bias[col];
#pragma unroll
    for (int r = 0; r < 8; ++r)
      Z[(long)(r0 + hi * 8 + r) * D + col] = acc[nt][r] + bv;
  }
}

// Combine: Z = [h0 h1 h2] @ Wc + b, K = 384 spanning the 3 branch buffers.
__global__ void __launch_bounds__(256)
gemm_comb(const bf16* __restrict__ A /* [3][N][128] */, const bf16* __restrict__ wpack,
          const float* __restrict__ bias, float* __restrict__ Z) {
  const int lane = threadIdx.x & 31;
  const int wv   = threadIdx.x >> 5;
  const int r0   = blockIdx.x * 128 + wv * 16;
  const int m    = lane & 15;
  const int hi   = lane >> 4;
  const int dofs = hi << 3;
  const long row = r0 + m;

  v8f acc[8];
#pragma unroll
  for (int t = 0; t < 8; ++t) acc[t] = v8f{};
#pragma unroll
  for (int ks = 0; ks < 12; ++ks) {
    const int brk = ks >> 2;
    const int kin = (ks & 3) * 32;
    v16bf a = load_a_bf16(A + ((long)brk * N_CELLS + row) * D + kin + dofs);
#pragma unroll
    for (int nt = 0; nt < 8; ++nt) {
      v16bf b = *(const v16bf*)(wpack + (((ks << 3) + nt) << 9) + (lane << 4));
      acc[nt] = __builtin_amdgcn_wmma_f32_16x16x32_bf16(
          false, a, false, b, (short)0, acc[nt], false, false);
    }
  }
  const int col0 = lane & 15;
#pragma unroll
  for (int nt = 0; nt < 8; ++nt) {
    int col = col0 + (nt << 4);
    float bv = bias[col];
#pragma unroll
    for (int r = 0; r < 8; ++r)
      Z[(long)(r0 + hi * 8 + r) * D + col] = acc[nt][r] + bv;
  }
}

// ---------------------------------------------------------------------------
// BatchNorm over the node axis: per-column sum / sumsq (coalesced, LDS reduce,
// atomics into small stats buffers), then elementwise normalize + ReLU.
// ---------------------------------------------------------------------------
__global__ void bn_stats(const float* __restrict__ Z, float* __restrict__ sum,
                         float* __restrict__ sq) {
  const int br = blockIdx.y;
  const float* base = Z + (long)br * N_CELLS * D + (long)blockIdx.x * 256 * D;
  const int col  = threadIdx.x & 127;
  const int half = threadIdx.x >> 7;
  float s = 0.f, q = 0.f;
  for (int r = half; r < 256; r += 2) {
    float v = base[(long)r * D + col];
    s += v; q += v * v;
  }
  __shared__ float ls[256], lq[256];
  ls[threadIdx.x] = s; lq[threadIdx.x] = q;
  __syncthreads();
  if (threadIdx.x < 128) {
    s = ls[threadIdx.x] + ls[threadIdx.x + 128];
    q = lq[threadIdx.x] + lq[threadIdx.x + 128];
    atomAddF(&sum[br * D + col], s);
    atomAddF(&sq[br * D + col], q);
  }
}

__global__ void bn_norm_bf16(const float* __restrict__ Z, const float* __restrict__ sum,
                             const float* __restrict__ sq, const float* __restrict__ g,
                             const float* __restrict__ be, bf16* __restrict__ out) {
  const int br = blockIdx.y;
  long gi = (long)blockIdx.x * 256 + threadIdx.x;   // over N*D
  int col = (int)(gi & 127);
  int bc2 = br * D + col;
  long off = (long)br * N_CELLS * D + gi;
  float mean = sum[bc2] * INV_N;
  float var  = sq[bc2] * INV_N - mean * mean;
  float inv  = rsqrtf(var + BN_EPS);
  float v = (Z[off] - mean) * inv * g[bc2] + be[bc2];
  v = v > 0.f ? v : 0.f;
  out[off] = f2bf(v);
}

__global__ void bn_norm_f32(const float* __restrict__ Z, const float* __restrict__ sum,
                            const float* __restrict__ sq, const float* __restrict__ g,
                            const float* __restrict__ be, float* __restrict__ out) {
  long gi = (long)blockIdx.x * 256 + threadIdx.x;   // over N*D
  int col = (int)(gi & 127);
  float mean = sum[col] * INV_N;
  float var  = sq[col] * INV_N - mean * mean;
  float inv  = rsqrtf(var + BN_EPS);
  float v = (Z[gi] - mean) * inv * g[col] + be[col];
  out[gi] = v > 0.f ? v : 0.f;
}

// ---------------------------------------------------------------------------
extern "C" void kernel_launch(void* const* d_in, const int* in_sizes, int n_in,
                              void* d_out, int out_size, void* d_ws, size_t ws_size,
                              hipStream_t stream) {
  (void)in_sizes; (void)n_in; (void)out_size; (void)ws_size;
  const float* x       = (const float*)d_in[0];
  const int*   up_idx  = (const int*)d_in[1];
  const int*   dn_idx  = (const int*)d_in[2];
  const int*   b_idx   = (const int*)d_in[3];
  const float* up_attr = (const float*)d_in[4];
  const float* dn_attr = (const float*)d_in[5];
  const float* b_attr  = (const float*)d_in[6];
  const float* w_up    = (const float*)d_in[7];
  const float* b_up    = (const float*)d_in[8];
  const float* w_dn    = (const float*)d_in[9];
  const float* b_dn    = (const float*)d_in[10];
  const float* w1      = (const float*)d_in[11];
  const float* b1      = (const float*)d_in[12];
  const float* g1      = (const float*)d_in[13];
  const float* be1     = (const float*)d_in[14];
  const float* w2      = (const float*)d_in[15];
  const float* b2      = (const float*)d_in[16];
  const float* g2      = (const float*)d_in[17];
  const float* be2     = (const float*)d_in[18];
  const float* wc      = (const float*)d_in[19];
  const float* cb_b    = (const float*)d_in[20];
  const float* cb_g    = (const float*)d_in[21];
  const float* cb_be   = (const float*)d_in[22];
  float* out = (float*)d_out;

  const long SZF = 3L * N_CELLS * D;     // 12.58M elems per f32 triple-buffer
  char* ws = (char*)d_ws;
  float* accH  = (float*)ws; ws += SZF * 4;        // messages + x ; later z2
  float* zbuf  = (float*)ws; ws += SZF * 4;        // z1 ; later combine z
  bf16*  hb    = (bf16*) ws; ws += SZF * 2;        // bf16 activations (reused)
  float* stats = (float*)ws; ws += 1792 * 4;       // sums/sumsq, zeroed each call
  bf16*  wupP  = (bf16*) ws; ws += 32768 * 2;
  bf16*  wdnP  = (bf16*) ws; ws += 32768 * 2;
  bf16*  w1P   = (bf16*) ws; ws += 3 * 16384 * 2;
  bf16*  w2P   = (bf16*) ws; ws += 3 * 16384 * 2;
  bf16*  wcP   = (bf16*) ws; ws += 49152 * 2;

  float* sum1 = stats;        float* sq1 = stats + 384;
  float* sum2 = stats + 768;  float* sq2 = stats + 1152;
  float* sumc = stats + 1536; float* sqc = stats + 1664;

  dim3 blk(256);
  // weight packing (deterministic, re-done every call)
  pack_w<<<dim3(128, 1), blk, 0, stream>>>(w_up, wupP, 256);
  pack_w<<<dim3(128, 1), blk, 0, stream>>>(w_dn, wdnP, 256);
  pack_w<<<dim3(64, 3),  blk, 0, stream>>>(w1, w1P, 128);
  pack_w<<<dim3(64, 3),  blk, 0, stream>>>(w2, w2P, 128);
  pack_w<<<dim3(192, 1), blk, 0, stream>>>(wc, wcP, 384);

  // acc = x (x3 branches), stats = 0
  init_acc<<<dim3(N_CELLS * D / 256), blk, 0, stream>>>(x, accH, stats);

  // message passing (WMMA gathered GEMM + atomic scatter)
  msg_kernel<<<dim3(E_UP / 128), blk, 0, stream>>>(x, up_idx, up_attr, wupP, b_up,
                                                   accH, E_UP);
  msg_kernel<<<dim3(E_DN / 128), blk, 0, stream>>>(x, dn_idx, dn_attr, wdnP, b_dn,
                                                   accH + (long)N_CELLS * D, E_DN);
  boundary_kernel<<<dim3(E_B * 32 / 256), blk, 0, stream>>>(b_idx, b_attr,
                                                   accH + 2L * N_CELLS * D);

  // update layer 1: GEMM -> BN stats -> normalize+ReLU (to bf16)
  gemm_f32A<<<dim3(N_CELLS / 128, 3), blk, 0, stream>>>(accH, w1P, b1, zbuf);
  bn_stats<<<dim3(N_CELLS / 256, 3), blk, 0, stream>>>(zbuf, sum1, sq1);
  bn_norm_bf16<<<dim3(N_CELLS * D / 256, 3), blk, 0, stream>>>(zbuf, sum1, sq1, g1, be1, hb);

  // update layer 2 (z2 overwrites accH)
  gemm_bf16A<<<dim3(N_CELLS / 128, 3), blk, 0, stream>>>(hb, w2P, b2, accH);
  bn_stats<<<dim3(N_CELLS / 256, 3), blk, 0, stream>>>(accH, sum2, sq2);
  bn_norm_bf16<<<dim3(N_CELLS * D / 256, 3), blk, 0, stream>>>(accH, sum2, sq2, g2, be2, hb);

  // combine (z into zbuf), final BN + ReLU to d_out (f32)
  gemm_comb<<<dim3(N_CELLS / 128, 1), blk, 0, stream>>>(hb, wcP, cb_b, zbuf);
  bn_stats<<<dim3(N_CELLS / 256, 1), blk, 0, stream>>>(zbuf, sumc, sqc);
  bn_norm_f32<<<dim3(N_CELLS * D / 256, 1), blk, 0, stream>>>(zbuf, sumc, sqc, cb_g, cb_be, out);
}
